// MultiHeadSelfAttention_13529146983109
// MI455X (gfx1250) — compile-verified, matrix operand order checked by reading, not run on hardware
//
#include <hip/hip_runtime.h>
#include <math.h>

typedef _Float16 f16;
typedef __attribute__((ext_vector_type(16))) _Float16 v16h;
typedef __attribute__((ext_vector_type(8)))  _Float16 v8h;
typedef __attribute__((ext_vector_type(4)))  _Float16 v4h;
typedef __attribute__((ext_vector_type(8)))  float    v8f;
typedef __attribute__((ext_vector_type(4)))  unsigned int u32x4;
typedef __attribute__((ext_vector_type(8)))  int      i32x8;
typedef __attribute__((ext_vector_type(4)))  int      i32x4;

#define SEQ    2048
#define DMODEL 1024
#define NHEAD  16
#define DHEAD  64
#define MTOT   (2 * SEQ)     // 4096 = B*S
#define LDSK   72            // padded LDS row stride (144 B: 16B-aligned, bank-spread)

#define ASYNC_B128(ldsaddr, gaddr)                                             \
    asm volatile("global_load_async_to_lds_b128 %0, %1, off"                   \
                 :: "v"(ldsaddr), "v"(gaddr) : "memory")
#define WAIT_ASYNC() asm volatile("s_wait_asynccnt 0x0" ::: "memory")

__device__ __forceinline__ uint32_t lds_off(const void* p) {
    return (uint32_t)(size_t)p;   // flat LDS addr: low 32 bits = LDS offset
}

// ---------------------------------------------------------------------------
// TDM: 2-D tile (tile_x elems of 2B  x  tile_y rows) global -> LDS, with LDS
// row padding so rows land LDSK (=72) f16 apart.  Issue from ONE wave only.
// D# layout per CDNA5 ISA ch.8 (group0 128b, group1 256b; groups 2/3 zero).
// ---------------------------------------------------------------------------
__device__ __forceinline__ void tdm_load_2d_f16(uint32_t lds_byte_off, const void* gaddr,
                                                uint32_t tile_x, uint32_t tile_y,
                                                uint32_t tensor_x, uint32_t tensor_y,
                                                uint32_t row_stride_elems) {
    uint64_t ga = (uint64_t)gaddr;
    u32x4 g0;
    g0[0] = 1u;                                            // count=1, user D#
    g0[1] = lds_byte_off;                                  // lds_addr
    g0[2] = (uint32_t)(ga & 0xffffffffu);                  // global_addr[31:0]
    g0[3] = (uint32_t)((ga >> 32) & 0x01ffffffu) | (2u << 30);  // addr[56:32] | type=2
    i32x8 g1;
    g1[0] = (int)((1u << 16)        // data_size = 2 bytes
                | (1u << 20)        // pad_enable
                | (4u << 22)        // pad_interval: 32 DWORDs (=128B row)
                | (3u << 25));      // pad_amount: 4 DWORDs (=16B -> stride 144B)
    g1[1] = (int)((tensor_x & 0xffffu) << 16);                            // dim0[15:0]
    g1[2] = (int)(((tensor_x >> 16) & 0xffffu) | ((tensor_y & 0xffffu) << 16));
    g1[3] = (int)(((tensor_y >> 16) & 0xffffu) | ((tile_x & 0xffffu) << 16));
    g1[4] = (int)(tile_y & 0xffffu);                                      // tile_dim1 (dim2=0)
    g1[5] = (int)row_stride_elems;                                        // dim0_stride[31:0]
    g1[6] = 0;
    g1[7] = 0;
    i32x4 z4 = {0, 0, 0, 0};
#if defined(__clang_major__) && (__clang_major__ >= 23)
    i32x8 z8 = {0, 0, 0, 0, 0, 0, 0, 0};
    __builtin_amdgcn_tensor_load_to_lds(g0, g1, z4, z4, z8, 0);
#else
    __builtin_amdgcn_tensor_load_to_lds(g0, g1, z4, z4, 0);
#endif
}
#define WAIT_TENSOR() __builtin_amdgcn_s_wait_tensorcnt(0)

union FragU { v16h v; v8h h[2]; };

// 16x32 f16 WMMA A/B fragment from LDS (row-major, `stride` elems).
__device__ __forceinline__ v16h load_frag(const f16* base, int stride) {
    int lane = threadIdx.x & 31;
    int m    = lane & 15;
    int half = lane >> 4;
    const f16* p = base + m * stride + half * 8;
    FragU f;
    f.h[0] = *(const v8h*)(p);        // K = half*8 .. +7
    f.h[1] = *(const v8h*)(p + 16);   // K = 16+half*8 .. +7
    return f.v;
}

__device__ __forceinline__ v8f wmma16(v16h a, v16h b, v8f c) {
    return __builtin_amdgcn_wmma_f32_16x16x32_f16(false, a, false, b, (short)0, c,
                                                  false, false);
}

// ---------------------------------------------------------------------------
// One-time converts / re-layouts (memory-bound prep, ~40 MB total)
// ---------------------------------------------------------------------------
__global__ __launch_bounds__(256)
void convert_f32_f16(const float* __restrict__ src, f16* __restrict__ dst, int n) {
    int i = (blockIdx.x * 256 + threadIdx.x) * 4;
    if (i < n) {
        float4 v = *(const float4*)(src + i);
        v4h o = { (f16)v.x, (f16)v.y, (f16)v.z, (f16)v.w };
        *(v4h*)(dst + i) = o;
    }
}

// Wt[N][K] = (f16) W[K][N]   (tiled via LDS so both sides stay coalesced)
__global__ __launch_bounds__(256)
void transpose_w(const float* __restrict__ W, f16* __restrict__ Wt, int K, int N) {
    __shared__ float T[64 * 65];
    int n0 = blockIdx.x * 64, k0 = blockIdx.y * 64;
    for (int i = threadIdx.x; i < 64 * 64; i += 256) {
        int r = i >> 6, c = i & 63;          // r = k, c = n
        T[r * 65 + c] = W[(size_t)(k0 + r) * N + n0 + c];
    }
    __syncthreads();
    for (int i = threadIdx.x; i < 64 * 64; i += 256) {
        int r = i >> 6, c = i & 63;          // r = n, c = k
        Wt[(size_t)(n0 + r) * K + k0 + c] = (f16)T[c * 65 + r];
    }
}

// Vt[b][h][d][s] = qkv16[b][s][2048 + h*64 + d]
__global__ __launch_bounds__(256)
void build_vt(const f16* __restrict__ qkv, f16* __restrict__ vt) {
    __shared__ f16 T[64 * LDSK];
    int s0 = blockIdx.x * 64, h = blockIdx.y, b = blockIdx.z;
    const f16* src = qkv + (size_t)b * SEQ * (3 * DMODEL) + 2 * DMODEL + h * DHEAD;
    for (int i = threadIdx.x; i < 64 * 64; i += 256) {
        int r = i >> 6, d = i & 63;          // r = s
        T[r * LDSK + d] = src[(size_t)(s0 + r) * (3 * DMODEL) + d];
    }
    __syncthreads();
    f16* dst = vt + (size_t)(b * NHEAD + h) * DHEAD * SEQ + s0;
    for (int i = threadIdx.x; i < 64 * 64; i += 256) {
        int d = i >> 6, c = i & 63;          // c = s offset
        dst[(size_t)d * SEQ + c] = T[c * LDSK + d];
    }
}

// ---------------------------------------------------------------------------
// C[M,N] = A[M,K] @ Bt[N,K]^T + bias  (A,Bt f16; out f16 or f32)
// 256 threads / 8 waves; tile 64x128, BK=64, double-buffered.
// A tile staged by TDM (wave 0, TENSORcnt); Bt tile by per-lane async (ASYNCcnt).
// ---------------------------------------------------------------------------
template <bool OUT16>
__global__ __launch_bounds__(256)
void gemm_async(const f16* __restrict__ A, const f16* __restrict__ Bt,
                const float* __restrict__ bias, void* __restrict__ Cout,
                int M, int N, int K) {
    __shared__ f16 As[2][64 * LDSK];     // 18.4 KB
    __shared__ f16 Bs[2][128 * LDSK];    // 36.9 KB
    const int tid = threadIdx.x, lane = tid & 31;
    const int w = tid >> 5, wr = w >> 2, wc = w & 3;
    const int m0 = blockIdx.y * 64, n0 = blockIdx.x * 128;

    auto issue = [&](int k0, int buf) {
        if (w == 0)   // one TDM per tile: 64 rows x 64 f16, row stride K, pad->LDSK
            tdm_load_2d_f16(lds_off(&As[buf][0]), A + (size_t)m0 * K + k0,
                            64, 64, (uint32_t)K, (uint32_t)M, (uint32_t)K);
        #pragma unroll
        for (int j = 0; j < 4; ++j) {                 // Bt: 1024 16B-chunks
            int c = tid + j * 256, r = c >> 3, c8 = c & 7;
            ASYNC_B128(lds_off(&Bs[buf][r * LDSK + c8 * 8]),
                       Bt + (size_t)(n0 + r) * K + k0 + c8 * 8);
        }
    };

    v8f acc[2][2];
    #pragma unroll
    for (int i = 0; i < 2; ++i)
        #pragma unroll
        for (int j = 0; j < 2; ++j) acc[i][j] = (v8f){0.f,0.f,0.f,0.f,0.f,0.f,0.f,0.f};

    issue(0, 0);
    const int nk = K / 64;
    for (int t = 0; t < nk; ++t) {
        WAIT_ASYNC();            // own async copies for buf cur done
        WAIT_TENSOR();           // wave0's TDM done (no-op elsewhere)
        __syncthreads();         // all waves: cur staged, prev compute retired
        if (t + 1 < nk) issue((t + 1) * 64, (t + 1) & 1);
        const f16* Ab = As[t & 1];
        const f16* Bb = Bs[t & 1];
        #pragma unroll
        for (int kk = 0; kk < 64; kk += 32) {
            v16h a0 = load_frag(Ab + (wr * 32 +  0) * LDSK + kk, LDSK);
            v16h a1 = load_frag(Ab + (wr * 32 + 16) * LDSK + kk, LDSK);
            v16h b0 = load_frag(Bb + (wc * 32 +  0) * LDSK + kk, LDSK);
            v16h b1 = load_frag(Bb + (wc * 32 + 16) * LDSK + kk, LDSK);
            acc[0][0] = wmma16(a0, b0, acc[0][0]);
            acc[0][1] = wmma16(a0, b1, acc[0][1]);
            acc[1][0] = wmma16(a1, b0, acc[1][0]);
            acc[1][1] = wmma16(a1, b1, acc[1][1]);
        }
    }

    const int half = lane >> 4, nlo = lane & 15;
    #pragma unroll
    for (int ti = 0; ti < 2; ++ti)
        #pragma unroll
        for (int tj = 0; tj < 2; ++tj) {
            int rbase = m0 + wr * 32 + ti * 16 + half * 8;
            int ccol  = n0 + wc * 32 + tj * 16 + nlo;
            float bv = bias[ccol];
            #pragma unroll
            for (int i = 0; i < 8; ++i) {
                float v = acc[ti][tj][i] + bv;
                if (OUT16) ((f16*)Cout)[(size_t)(rbase + i) * N + ccol] = (f16)v;
                else       ((float*)Cout)[(size_t)(rbase + i) * N + ccol] = v;
            }
        }
}

// ---------------------------------------------------------------------------
// Flash attention. Block = (b, h, 64 q-rows), 128 threads / 4 waves.
// qkv f16 [B,S,3D]; vt f16 [B,H,Dh,S]; out attn f16 [B,S,D].
// K staged via per-lane async (ASYNCcnt); V via TDM (wave0, TENSORcnt);
// double-buffered; f32 online softmax; WMMA f16.
// ---------------------------------------------------------------------------
__global__ __launch_bounds__(128)
void attn_async(const f16* __restrict__ qkv, const f16* __restrict__ vt,
                f16* __restrict__ attn) {
    __shared__ f16 Qs[64 * LDSK];        // [qrow][d]
    __shared__ f16 Ks[2][64 * LDSK];     // [key][d]  (Bt layout for QK^T)
    __shared__ f16 Vs[2][64 * LDSK];     // [d][key]  (Bt layout for PV)
    __shared__ f16 Ps[64 * LDSK];        // [qrow][key]

    const int tid = threadIdx.x, lane = tid & 31;
    const int w = tid >> 5, half = lane >> 4, nlo = lane & 15;
    const int q0 = blockIdx.x * 64, h = blockIdx.y, b = blockIdx.z;
    const f16* qb = qkv + (size_t)b * SEQ * (3 * DMODEL);
    const f16* vb = vt + (size_t)(b * NHEAD + h) * DHEAD * SEQ;
    const int hoff = h * DHEAD;

    // Q tile: 512 16B-chunks, 4 per thread (async)
    #pragma unroll
    for (int j = 0; j < 4; ++j) {
        int c = tid + j * 128, r = c >> 3, c8 = c & 7;
        ASYNC_B128(lds_off(&Qs[r * LDSK + c8 * 8]),
                   qb + (size_t)(q0 + r) * (3 * DMODEL) + hoff + c8 * 8);
    }
    auto issueKV = [&](int kb, int buf) {
        #pragma unroll
        for (int j = 0; j < 4; ++j) {   // K: async
            int c = tid + j * 128, r = c >> 3, c8 = c & 7;
            ASYNC_B128(lds_off(&Ks[buf][r * LDSK + c8 * 8]),
                       qb + (size_t)(kb + r) * (3 * DMODEL) + DMODEL + hoff + c8 * 8);
        }
        if (w == 0)                     // V: one TDM (64 rows x 64 f16, stride SEQ)
            tdm_load_2d_f16(lds_off(&Vs[buf][0]), vb + kb,
                            64, 64, (uint32_t)SEQ, (uint32_t)DHEAD, (uint32_t)SEQ);
    };
    issueKV(0, 0);

    float mrow[8], lrow[8];
    v8f o[4];
    #pragma unroll
    for (int i = 0; i < 8; ++i) { mrow[i] = -1e30f; lrow[i] = 0.f; }
    #pragma unroll
    for (int t = 0; t < 4; ++t) o[t] = (v8f){0.f,0.f,0.f,0.f,0.f,0.f,0.f,0.f};
    const float scale = 0.125f;   // 1/sqrt(64)

    const int nb = SEQ / 64;
    for (int t = 0; t < nb; ++t) {
        WAIT_ASYNC();
        WAIT_TENSOR();
        __syncthreads();
        if (t + 1 < nb) issueKV((t + 1) * 64, (t + 1) & 1);
        const f16* Kb = Ks[t & 1];
        const f16* Vb = Vs[t & 1];

        // S(16x64) = Q @ K^T
        v8f s[4];
        #pragma unroll
        for (int n = 0; n < 4; ++n) s[n] = (v8f){0.f,0.f,0.f,0.f,0.f,0.f,0.f,0.f};
        #pragma unroll
        for (int kk = 0; kk < 64; kk += 32) {
            v16h aq = load_frag(Qs + (w * 16) * LDSK + kk, LDSK);
            #pragma unroll
            for (int n = 0; n < 4; ++n)
                s[n] = wmma16(aq, load_frag(Kb + (n * 16) * LDSK + kk, LDSK), s[n]);
        }

        // online softmax; slot i <-> row half*8+i; xor-shuffles stay within halves
        #pragma unroll
        for (int i = 0; i < 8; ++i) {
            #pragma unroll
            for (int n = 0; n < 4; ++n) s[n][i] *= scale;
            float mx = fmaxf(fmaxf(s[0][i], s[1][i]), fmaxf(s[2][i], s[3][i]));
            #pragma unroll
            for (int msk = 1; msk < 16; msk <<= 1) mx = fmaxf(mx, __shfl_xor(mx, msk));
            float mnew = fmaxf(mrow[i], mx);
            float corr = __expf(mrow[i] - mnew);
            mrow[i] = mnew;
            float rs = 0.f;
            #pragma unroll
            for (int n = 0; n < 4; ++n) {
                float p = __expf(s[n][i] - mnew);
                s[n][i] = p;
                rs += p;
            }
            #pragma unroll
            for (int msk = 1; msk < 16; msk <<= 1) rs += __shfl_xor(rs, msk);
            lrow[i] = lrow[i] * corr + rs;
            #pragma unroll
            for (int n = 0; n < 4; ++n) o[n][i] *= corr;
        }

        // P: C-layout regs -> LDS row-major (wave-private rows; per-wave DS
        // ordering makes the store->frag-load below safe without a barrier)
        #pragma unroll
        for (int n = 0; n < 4; ++n)
            #pragma unroll
            for (int i = 0; i < 8; ++i)
                Ps[(w * 16 + half * 8 + i) * LDSK + n * 16 + nlo] = (f16)s[n][i];

        // O += P @ V
        #pragma unroll
        for (int kk = 0; kk < 64; kk += 32) {
            v16h ap = load_frag(Ps + (w * 16) * LDSK + kk, LDSK);
            #pragma unroll
            for (int n = 0; n < 4; ++n)
                o[n] = wmma16(ap, load_frag(Vb + (n * 16) * LDSK + kk, LDSK), o[n]);
        }
    }

    #pragma unroll
    for (int n = 0; n < 4; ++n)
        #pragma unroll
        for (int i = 0; i < 8; ++i) {
            int r = q0 + w * 16 + half * 8 + i;
            int c = hoff + n * 16 + nlo;
            attn[(size_t)b * SEQ * DMODEL + (size_t)r * DMODEL + c] =
                (f16)(o[n][i] / lrow[i]);
        }
}

// ---------------------------------------------------------------------------
extern "C" void kernel_launch(void* const* d_in, const int* in_sizes, int n_in,
                              void* d_out, int out_size, void* d_ws, size_t ws_size,
                              hipStream_t stream) {
    (void)in_sizes; (void)n_in; (void)out_size; (void)ws_size;
    const float* x     = (const float*)d_in[0];   // [2,2048,1024]
    const float* W_qkv = (const float*)d_in[1];   // [1024,3072]
    const float* b_qkv = (const float*)d_in[2];   // [3072]
    const float* W_out = (const float*)d_in[3];   // [1024,1024]
    const float* b_out = (const float*)d_in[4];   // [1024]
    float* out = (float*)d_out;

    // f16 workspace layout (element offsets), total ~58.7 MB
    f16* x16    = (f16*)d_ws;
    f16* wqkvT  = x16   + (size_t)MTOT * DMODEL;
    f16* woutT  = wqkvT + (size_t)(3 * DMODEL) * DMODEL;
    f16* qkv16  = woutT + (size_t)DMODEL * DMODEL;
    f16* vt16   = qkv16 + (size_t)MTOT * 3 * DMODEL;
    f16* attn16 = vt16  + (size_t)2 * NHEAD * DHEAD * SEQ;

    const int nx = MTOT * DMODEL;
    convert_f32_f16<<<nx / 1024, 256, 0, stream>>>(x, x16, nx);
    transpose_w<<<dim3((3 * DMODEL) / 64, DMODEL / 64), 256, 0, stream>>>(
        W_qkv, wqkvT, DMODEL, 3 * DMODEL);
    transpose_w<<<dim3(DMODEL / 64, DMODEL / 64), 256, 0, stream>>>(
        W_out, woutT, DMODEL, DMODEL);

    gemm_async<true><<<dim3((3 * DMODEL) / 128, MTOT / 64), 256, 0, stream>>>(
        x16, wqkvT, b_qkv, qkv16, MTOT, 3 * DMODEL, DMODEL);

    build_vt<<<dim3(SEQ / 64, NHEAD, 2), 256, 0, stream>>>(qkv16, vt16);

    attn_async<<<dim3(SEQ / 64, NHEAD, 2), 128, 0, stream>>>(qkv16, vt16, attn16);

    gemm_async<false><<<dim3(DMODEL / 128, MTOT / 64), 256, 0, stream>>>(
        attn16, woutT, b_out, out, MTOT, DMODEL, DMODEL);
}
